// MultiHeadAttention_14482629722890
// MI455X (gfx1250) — matrix operand order repaired and checked by reading, not compiled, not run on hardware
//
#include <hip/hip_runtime.h>
#include <hip/hip_bf16.h>

#define BB 4
#define SS 2048
#define DD 1024
#define HH 16
#define DHN 64
#define MM (BB * SS)        // 8192 rows in flattened [B*S, D]
#define PSTRIDE 2052        // padded LDS row stride (floats): 2052 % 64 == 4

typedef __attribute__((ext_vector_type(16))) __bf16 v16bf;
typedef __attribute__((ext_vector_type(8)))  float  v8f;

// ---------------------------------------------------------------- helpers ---

__device__ __forceinline__ unsigned short f2bf(float f) {
    unsigned int u = __float_as_uint(f);
    u += 0x7FFFu + ((u >> 16) & 1u);          // round-to-nearest-even
    return (unsigned short)(u >> 16);
}

// A-fragment, 16x32 bf16: lane holds 16 contiguous K elements starting at
// klo = (lane>>4)*16. 'p' already includes + klo.
__device__ __forceinline__ v16bf load_frag_a_bf16(const unsigned short* p) {
    union { v16bf v; uint4 q[2]; } u;
    u.q[0] = *(const uint4*)(p);
    u.q[1] = *(const uint4*)(p + 8);
    return u.v;
}

// B-fragment, 32x16 bf16: lane holds 16 contiguous K elements starting at
// (lane>>4)*16; 'p' already includes that offset. Two b128 loads.
__device__ __forceinline__ v16bf load_frag_b_bf16(const unsigned short* p) {
    union { v16bf v; uint4 q[2]; } u;
    u.q[0] = *(const uint4*)(p);
    u.q[1] = *(const uint4*)(p + 8);
    return u.v;
}

// A-fragment built from f32 source (LDS-resident probabilities), converted
// on the fly to bf16.
__device__ __forceinline__ v16bf load_frag_a_f32(const float* p) {
    union { v16bf v; unsigned short s[16]; } u;
#pragma unroll
    for (int i = 0; i < 8; ++i) {
        u.s[i]     = f2bf(p[i]);
        u.s[8 + i] = f2bf(p[8 + i]);
    }
    return u.v;
}

__device__ __forceinline__ v8f wmma_bf16(v16bf a, v16bf b, v8f c) {
    // (neg_a, A, neg_b, B, c_mod, C, reuse_a, reuse_b)
    return __builtin_amdgcn_wmma_f32_16x16x32_bf16(false, a, false, b,
                                                   (short)0, c, false, false);
}

// ------------------------------------------------------------- f32 -> bf16 --

__global__ __launch_bounds__(256) void cvt_f32_bf16(const float* __restrict__ in,
                                                    unsigned short* __restrict__ out,
                                                    int n4) {
    int i = blockIdx.x * 256 + threadIdx.x;
    if (i >= n4) return;
    float4 f = ((const float4*)in)[i];
    union { unsigned short s[4]; uint2 u; } o;
    o.s[0] = f2bf(f.x); o.s[1] = f2bf(f.y);
    o.s[2] = f2bf(f.z); o.s[3] = f2bf(f.w);
    ((uint2*)out)[i] = o.u;
}

// -------------------------------------------------- C = A[M,K] * B[N,K]^T ---
// Wave-tile GEMM: each wave computes a 16x64 strip of C (A-fragment reused
// across 4 WMMAs per 32-wide K step). 8 waves / block.

enum { OUT_BF16 = 0, OUT_F32 = 1, OUT_VT = 2 };

template <int OMODE>
__global__ __launch_bounds__(256) void gemm_anbt(const unsigned short* __restrict__ A,
                                                 const unsigned short* __restrict__ Bw,
                                                 void* __restrict__ Cout,
                                                 int M, int N, int Kd) {
    const int lane = threadIdx.x & 31;
    const int wave = threadIdx.x >> 5;
    const int lh   = lane >> 4;      // half-wave select
    const int ll   = lane & 15;
    const int nStrips = N >> 6;
    const int w      = blockIdx.x * 8 + wave;
    const int mTile  = w / nStrips;
    const int nStrip = w % nStrips;
    if (mTile * 16 >= M) return;     // wave-uniform

    const unsigned short* aP  = A  + (size_t)(mTile * 16 + ll) * Kd + lh * 16;
    const unsigned short* bP0 = Bw + (size_t)(nStrip * 64 +  0 + ll) * Kd + lh * 16;
    const unsigned short* bP1 = Bw + (size_t)(nStrip * 64 + 16 + ll) * Kd + lh * 16;
    const unsigned short* bP2 = Bw + (size_t)(nStrip * 64 + 32 + ll) * Kd + lh * 16;
    const unsigned short* bP3 = Bw + (size_t)(nStrip * 64 + 48 + ll) * Kd + lh * 16;

    v8f c0 = {}, c1 = {}, c2 = {}, c3 = {};
    for (int kk = 0; kk < Kd; kk += 32) {
        __builtin_prefetch(aP + kk + 64, 0, 1);
        v16bf a  = load_frag_a_bf16(aP  + kk);
        v16bf b0 = load_frag_b_bf16(bP0 + kk);
        v16bf b1 = load_frag_b_bf16(bP1 + kk);
        v16bf b2 = load_frag_b_bf16(bP2 + kk);
        v16bf b3 = load_frag_b_bf16(bP3 + kk);
        c0 = wmma_bf16(a, b0, c0);
        c1 = wmma_bf16(a, b1, c1);
        c2 = wmma_bf16(a, b2, c2);
        c3 = wmma_bf16(a, b3, c3);
    }

#pragma unroll
    for (int r = 0; r < 8; ++r) {
        const int m = mTile * 16 + r + 8 * lh;
        float v[4] = { c0[r], c1[r], c2[r], c3[r] };
#pragma unroll
        for (int t = 0; t < 4; ++t) {
            const int n = nStrip * 64 + t * 16 + ll;
            if (OMODE == OUT_F32) {
                ((float*)Cout)[(size_t)m * N + n] = v[t];
            } else if (OMODE == OUT_BF16) {
                ((unsigned short*)Cout)[(size_t)m * N + n] = f2bf(v[t]);
            } else {  // OUT_VT: scatter V into [B,H,DH,S] for transpose-free PV
                const int b  = m >> 11;          // m / S
                const int s  = m & (SS - 1);
                const int h  = n >> 6;
                const int dh = n & 63;
                ((unsigned short*)Cout)[((size_t)(b * HH + h) * DHN + dh) * SS + s] =
                    f2bf(v[t]);
            }
        }
    }
}

// --------------------------------------------------- fused causal attention --
// One workgroup (8 waves) per (b, h, 16-query tile). The 16x2048 f32 score
// strip lives entirely in LDS (CDNA5: 320 KB/WGP), so the 1 GiB attn region
// is written to HBM exactly once (normalized probs), instead of the
// write-raw / read / rewrite / re-read dance (2.5 GiB -> 1.0 GiB HBM).
//
// Phase 1: waves cooperatively compute causal score tiles -> LDS.
// Phase 2: per-row softmax over LDS; probs -> global attn (all cols, clears
//          poison) and back into LDS for cols < kAlign.
// Phase 3: PV with split-K across waves; A-fragments from LDS (f32->bf16),
//          B-fragments from V^T in global; ds_add_f32 reduction.
// Phase 4: bf16 context written for the O-projection GEMM.

__global__ __launch_bounds__(256) void attn_fused(const unsigned short* __restrict__ Q,
                                                  const unsigned short* __restrict__ Kc,
                                                  const unsigned short* __restrict__ VT,
                                                  float* __restrict__ attn,
                                                  unsigned short* __restrict__ Ctx) {
    extern __shared__ char smem_raw[];
    float* strip = (float*)smem_raw;           // 16 x PSTRIDE f32
    float* pvacc = strip + 16 * PSTRIDE;       // 16 x 64 f32 reduction buffer

    const int tid  = threadIdx.x;
    const int lane = tid & 31, wave = tid >> 5;
    const int lh = lane >> 4, ll = lane & 15;
    const int qt = blockIdx.x & 127;           // S/16 = 128 query tiles
    const int bh = blockIdx.x >> 7;
    const int b  = bh >> 4;
    const int h  = bh & 15;
    const int kAlign = (qt * 16 + 16 + 31) & ~31;   // K extent for PV (x32)

    // zero PV accumulators (256 threads x float4 == 16*64 floats)
    float4 z4; z4.x = 0.f; z4.y = 0.f; z4.z = 0.f; z4.w = 0.f;
    ((float4*)pvacc)[tid] = z4;

    // ---- phase 1: causal scores into LDS ----
    const unsigned short* qRow = Q + (size_t)(b * SS + qt * 16 + ll) * DD + h * DHN;
    v16bf a0 = load_frag_a_bf16(qRow + lh * 16);        // dh 0..31
    v16bf a1 = load_frag_a_bf16(qRow + 32 + lh * 16);   // dh 32..63
    for (int j = wave; j <= qt; j += 8) {              // wave-uniform bounds
        const unsigned short* kRow =
            Kc + (size_t)(b * SS + j * 16 + ll) * DD + h * DHN;
        v8f c = {};
        c = wmma_bf16(a0, load_frag_b_bf16(kRow + lh * 16), c);
        c = wmma_bf16(a1, load_frag_b_bf16(kRow + 32 + lh * 16), c);
        const int col = j * 16 + ll;
#pragma unroll
        for (int r = 0; r < 8; ++r)
            strip[(r + 8 * lh) * PSTRIDE + col] = c[r] * 0.125f;  // 1/sqrt(64)
    }
    __syncthreads();

    // ---- phase 2: softmax (2 rows per wave) ----
#pragma unroll
    for (int rr = 0; rr < 2; ++rr) {
        const int r = wave * 2 + rr;
        const int i = qt * 16 + r;                    // causal limit
        float* srow = strip + r * PSTRIDE;
        float mx = -3.402823466e38f;
        for (int c = lane; c <= i; c += 32) mx = fmaxf(mx, srow[c]);
#pragma unroll
        for (int o = 16; o > 0; o >>= 1) mx = fmaxf(mx, __shfl_xor(mx, o, 32));
        float sum = 0.0f;
        for (int c = lane; c <= i; c += 32) sum += __expf(srow[c] - mx);
#pragma unroll
        for (int o = 16; o > 0; o >>= 1) sum += __shfl_xor(sum, o, 32);
        const float inv = 1.0f / sum;
        float* grow = attn + ((size_t)bh * SS + i) * SS;
        for (int c = lane; c < SS; c += 32) {
            const float p = (c <= i) ? __expf(srow[c] - mx) * inv : 0.0f;
            grow[c] = p;                              // single HBM pass
            if (c < kAlign) srow[c] = p;              // PV source in LDS
        }
    }
    __syncthreads();

    // ---- phase 3: PV, split-K across waves ----
    const unsigned short* vt0 = VT + ((size_t)bh * DHN +  0 + ll) * SS + lh * 16;
    const unsigned short* vt1 = VT + ((size_t)bh * DHN + 16 + ll) * SS + lh * 16;
    const unsigned short* vt2 = VT + ((size_t)bh * DHN + 32 + ll) * SS + lh * 16;
    const unsigned short* vt3 = VT + ((size_t)bh * DHN + 48 + ll) * SS + lh * 16;

    v8f c0 = {}, c1 = {}, c2 = {}, c3 = {};
    bool did = false;                                  // wave-uniform
    for (int kk = wave * 32; kk < kAlign; kk += 256) {
        did = true;
        const float* pSrc = strip + ll * PSTRIDE + kk + lh * 16;
        v16bf a = load_frag_a_f32(pSrc);
        c0 = wmma_bf16(a, load_frag_b_bf16(vt0 + kk), c0);
        c1 = wmma_bf16(a, load_frag_b_bf16(vt1 + kk), c1);
        c2 = wmma_bf16(a, load_frag_b_bf16(vt2 + kk), c2);
        c3 = wmma_bf16(a, load_frag_b_bf16(vt3 + kk), c3);
    }
    if (did) {
#pragma unroll
        for (int r = 0; r < 8; ++r) {
            float* dst = pvacc + (r + 8 * lh) * 64 + ll;
            atomicAdd(dst +  0, c0[r]);
            atomicAdd(dst + 16, c1[r]);
            atomicAdd(dst + 32, c2[r]);
            atomicAdd(dst + 48, c3[r]);
        }
    }
    __syncthreads();

    // ---- phase 4: write bf16 context ----
#pragma unroll
    for (int i2 = 0; i2 < 4; ++i2) {
        const int idx = tid * 4 + i2;                 // 0..1023
        const int r = idx >> 6, n = idx & 63;
        Ctx[(size_t)(b * SS + qt * 16 + r) * DD + h * DHN + n] = f2bf(pvacc[idx]);
    }
}

// ------------------------------------------------------------------ launch ---

extern "C" void kernel_launch(void* const* d_in, const int* in_sizes, int n_in,
                              void* d_out, int out_size, void* d_ws, size_t ws_size,
                              hipStream_t stream) {
    const float* x  = (const float*)d_in[0];
    // d_in[1] = causal mask (triu, k=1): applied analytically, not read.
    const float* wq = (const float*)d_in[2];
    const float* wk = (const float*)d_in[3];
    const float* wv = (const float*)d_in[4];
    const float* wo = (const float*)d_in[5];

    float* out  = (float*)d_out;                    // [B,S,D]
    float* attn = out + (size_t)BB * SS * DD;       // [B,H,S,S]

    unsigned short* xb  = (unsigned short*)d_ws;    // x  bf16  [M,D]
    unsigned short* wqb = xb  + (size_t)MM * DD;    // weights bf16 [D,D]
    unsigned short* wkb = wqb + (size_t)DD * DD;
    unsigned short* wvb = wkb + (size_t)DD * DD;
    unsigned short* wob = wvb + (size_t)DD * DD;
    unsigned short* qb  = wob + (size_t)DD * DD;    // Q bf16 [B,S,H,DH]
    unsigned short* kb  = qb  + (size_t)MM * DD;    // K bf16 [B,S,H,DH]
    unsigned short* vtb = kb  + (size_t)MM * DD;    // V bf16 [B,H,DH,S]
    unsigned short* ctx = xb;                       // reuse xb after V-proj

    cvt_f32_bf16<<<(MM * DD / 4 + 255) / 256, 256, 0, stream>>>(x,  xb,  MM * DD / 4);
    cvt_f32_bf16<<<(DD * DD / 4 + 255) / 256, 256, 0, stream>>>(wq, wqb, DD * DD / 4);
    cvt_f32_bf16<<<(DD * DD / 4 + 255) / 256, 256, 0, stream>>>(wk, wkb, DD * DD / 4);
    cvt_f32_bf16<<<(DD * DD / 4 + 255) / 256, 256, 0, stream>>>(wv, wvb, DD * DD / 4);
    cvt_f32_bf16<<<(DD * DD / 4 + 255) / 256, 256, 0, stream>>>(wo, wob, DD * DD / 4);

    const int gemmBlocks = (MM / 16) * (DD / 64) / 8;      // 1024
    gemm_anbt<OUT_BF16><<<gemmBlocks, 256, 0, stream>>>(xb, wqb, qb,  MM, DD, DD);
    gemm_anbt<OUT_BF16><<<gemmBlocks, 256, 0, stream>>>(xb, wkb, kb,  MM, DD, DD);
    gemm_anbt<OUT_VT  ><<<gemmBlocks, 256, 0, stream>>>(xb, wvb, vtb, MM, DD, DD);

    const size_t ldsBytes = (16 * PSTRIDE + 16 * 64) * sizeof(float);  // 135,424
    attn_fused<<<BB * HH * (SS / 16), 256, ldsBytes, stream>>>(qb, kb, vtb, attn, ctx);

    gemm_anbt<OUT_F32><<<gemmBlocks, 256, 0, stream>>>(ctx, wob, out, MM, DD, DD);

    (void)in_sizes; (void)n_in; (void)out_size; (void)ws_size;
}
